// LongformerLayer_23235773071562
// MI455X (gfx1250) — matrix operand order repaired and checked by reading, not compiled, not run on hardware
//
#include <hip/hip_runtime.h>

// ---------------------------------------------------------------------------
// Longformer layer for MI455X (gfx1250, wave32, WMMA + Tensor Data Mover).
// All heavy math runs through v_wmma_f32_16x16x32_bf16 (f32 accumulate);
// GEMM tiles are staged into LDS by the TDM (tensor_load_to_lds, TENSORcnt).
// ---------------------------------------------------------------------------

typedef float        v8f   __attribute__((ext_vector_type(8)));
typedef __bf16       v16bf __attribute__((ext_vector_type(16)));
typedef __bf16       bf8v  __attribute__((ext_vector_type(8)));
typedef __bf16       bf4v  __attribute__((ext_vector_type(4)));
typedef unsigned int u32x4 __attribute__((ext_vector_type(4)));
typedef int          i32x4 __attribute__((ext_vector_type(4)));
typedef int          i32x8 __attribute__((ext_vector_type(8)));

#define DMODEL 1024
#define FFNDIM 4096
#define SEQ    4096
#define BATCH  2
#define NHEAD  16
#define HEADD  64
#define WIN    256
#define NBLK   (SEQ / WIN)
#define MROWS  (BATCH * SEQ)

// ---------------------------------------------------------------------------
// helpers
// ---------------------------------------------------------------------------
static __device__ __forceinline__ v8f wmma_bf16(v16bf a, v16bf b, v8f c) {
  // D = A(16x32 bf16) * B(32x16 bf16) + C(16x16 f32)
  return __builtin_amdgcn_wmma_f32_16x16x32_bf16(false, a, false, b, (short)0, c,
                                                 false, false);
}

static __device__ __forceinline__ v16bf combine16(bf8v lo, bf8v hi) {
  return __builtin_shufflevector(lo, hi, 0, 1, 2, 3, 4, 5, 6, 7,
                                 8, 9, 10, 11, 12, 13, 14, 15);
}

static __device__ __forceinline__ v8f v8_splat(float x) {
  v8f r;
#pragma unroll
  for (int i = 0; i < 8; ++i) r[i] = x;
  return r;
}

static __device__ __forceinline__ v8f v8_max(v8f a, v8f b) {
  v8f r;
#pragma unroll
  for (int i = 0; i < 8; ++i) r[i] = fmaxf(a[i], b[i]);
  return r;
}

// reductions across the 16-lane halves (lanes 0-15 / 16-31 independently),
// matching the WMMA C-fragment N striping.
static __device__ __forceinline__ v8f v8_half_reduce_max(v8f v) {
#pragma unroll
  for (int m = 1; m < 16; m <<= 1) {
    v8f o;
#pragma unroll
    for (int i = 0; i < 8; ++i) o[i] = __shfl_xor(v[i], m, 32);
    v = v8_max(v, o);
  }
  return v;
}

static __device__ __forceinline__ v8f v8_half_reduce_add(v8f v) {
#pragma unroll
  for (int m = 1; m < 16; m <<= 1) {
#pragma unroll
    for (int i = 0; i < 8; ++i) {
      float o = __shfl_xor(v[i], m, 32);
      v[i] += o;
    }
  }
  return v;
}

// ---------------------------------------------------------------------------
// Tensor Data Mover: async-load a 2D tile (128 rows x 32 bf16, row pitch
// k_elems) from global memory into LDS at lds_addr, inserting 16 bytes of
// padding after every 64-byte row -> LDS pitch of 40 bf16 (conflict-free
// fragment loads across all 64 banks). Tracked with TENSORcnt.
// D# layout per CDNA5 ISA.
// ---------------------------------------------------------------------------
static __device__ __forceinline__ void tdm_load_tile_2d(unsigned lds_addr,
                                                        const void* gptr,
                                                        unsigned k_elems) {
  const unsigned long long ga = (unsigned long long)(size_t)gptr;
  u32x4 g0;
  g0[0] = 1u;                                        // count=1 (valid user D#)
  g0[1] = lds_addr;                                  // lds_addr [63:32]
  g0[2] = (unsigned)ga;                              // global_addr[31:0]
  g0[3] = ((unsigned)(ga >> 32) & 0x01ffffffu) | (2u << 30);  // addr[56:32]|type=2
  i32x8 g1;
  // wg_mask=0 | data_size=1 (2B) | pad_enable | pad_interval=3 (16 DW = 64B)
  // | pad_amount=3 (4 DW = 16B)
  g1[0] = (1 << 16) | (1 << 20) | (3 << 22) | (3 << 25);
  g1[1] = (int)((k_elems & 0xffffu) << 16);          // tensor_dim0[15:0]
  g1[2] = (int)((k_elems >> 16) | (128u << 16));     // tensor_dim0[31:16]|dim1 lo
  g1[3] = (int)(32u << 16);                          // dim1 hi=0 | tile_dim0=32
  g1[4] = 128;                                       // tile_dim1=128, tile_dim2=0
  g1[5] = (int)k_elems;                              // tensor_dim0_stride lo
  g1[6] = 0;                                         // stride hi | dim1_stride lo
  g1[7] = 0;
  i32x4 gz = {0, 0, 0, 0};
#if defined(__clang_major__) && (__clang_major__ >= 23)
  i32x8 gz8 = {0, 0, 0, 0, 0, 0, 0, 0};
  __builtin_amdgcn_tensor_load_to_lds(g0, g1, gz, gz, gz8, 0);
#else
  __builtin_amdgcn_tensor_load_to_lds(g0, g1, gz, gz, 0);
#endif
}

static __device__ __forceinline__ unsigned lds_byte_addr(const void* p) {
  // LDS aperture: flat addr[31:0] is the byte offset within the WG's LDS.
  return (unsigned)(size_t)p;
}

// ---------------------------------------------------------------------------
// f32 -> bf16 elementwise convert (vectorized x4)
// ---------------------------------------------------------------------------
__global__ __launch_bounds__(256) void k_cvt_bf16(const float* __restrict__ src,
                                                  __bf16* __restrict__ dst,
                                                  int n4) {
  int i = blockIdx.x * 256 + threadIdx.x;
  if (i >= n4) return;
  float4 v = reinterpret_cast<const float4*>(src)[i];
  bf4v o;
  o[0] = (__bf16)v.x; o[1] = (__bf16)v.y; o[2] = (__bf16)v.z; o[3] = (__bf16)v.w;
  reinterpret_cast<bf4v*>(dst)[i] = o;
}

// ---------------------------------------------------------------------------
// W (K x N, f32, row major) -> Wt (N x K, bf16): B-operand layout for WMMA.
// ---------------------------------------------------------------------------
__global__ __launch_bounds__(256) void k_transpose_cvt(const float* __restrict__ src,
                                                       __bf16* __restrict__ dst,
                                                       int K, int N) {
  __shared__ float tile[32][33];
  const int n0 = blockIdx.x * 32, k0 = blockIdx.y * 32;
  const int tx = threadIdx.x, ty = threadIdx.y;  // 32 x 8
#pragma unroll
  for (int j = 0; j < 4; ++j)
    tile[ty + 8 * j][tx] = src[(size_t)(k0 + ty + 8 * j) * N + n0 + tx];
  __syncthreads();
#pragma unroll
  for (int j = 0; j < 4; ++j)
    dst[(size_t)(n0 + ty + 8 * j) * K + k0 + tx] = (__bf16)tile[tx][ty + 8 * j];
}

// ---------------------------------------------------------------------------
// Tiled WMMA GEMM: C(MxN) = A(MxK) @ Bt(NxK)^T + bias
//   128x128 workgroup tile, 8 waves (2x4), each wave 64x32 (4x2 wmma tiles).
//   A/B tiles staged to LDS by the TDM with double buffering: wave 0 issues
//   tensor_load_to_lds for tile kt+1 while all waves run WMMAs on tile kt.
// EPI: 0 = f32 store, 1 = bf16 store, 2 = bf16 transposed (for V^T), 3 = gelu bf16
// ---------------------------------------------------------------------------
template <int EPI>
__global__ __launch_bounds__(256) void k_gemm(const __bf16* __restrict__ A,
                                              const __bf16* __restrict__ Bt,
                                              const float* __restrict__ bias,
                                              float* __restrict__ Cf,
                                              __bf16* __restrict__ Cb,
                                              int M, int N, int K, int t_ld) {
  __shared__ __bf16 lA[2][128 * 40];
  __shared__ __bf16 lB[2][128 * 40];
  const int tid  = threadIdx.x;
  const int lane = tid & 31, wave = tid >> 5;
  const int wm = wave >> 2, wn = wave & 3;     // 2 x 4 wave grid
  const int lr = lane & 15, hx = lane >> 4;    // lane row / half select
  const bool issuer = (wave == 0);

  const __bf16* Abase = A  + (size_t)(blockIdx.y * 128) * K;
  const __bf16* Bbase = Bt + (size_t)(blockIdx.x * 128) * K;

  v8f acc[4][2];
#pragma unroll
  for (int mt = 0; mt < 4; ++mt)
#pragma unroll
    for (int nt = 0; nt < 2; ++nt) acc[mt][nt] = v8_splat(0.f);

  if (issuer) {  // prologue: TDM loads for tile 0
    tdm_load_tile_2d(lds_byte_addr(&lA[0][0]), Abase, (unsigned)K);
    tdm_load_tile_2d(lds_byte_addr(&lB[0][0]), Bbase, (unsigned)K);
  }

  const int ksteps = K >> 5;
  for (int kt = 0; kt < ksteps; ++kt) {
    const int buf = kt & 1;
    __syncthreads();  // readers of back buffer (iter kt-1) are done
    if (issuer) {
      if (kt + 1 < ksteps) {
        tdm_load_tile_2d(lds_byte_addr(&lA[buf ^ 1][0]), Abase + (kt + 1) * 32,
                         (unsigned)K);
        tdm_load_tile_2d(lds_byte_addr(&lB[buf ^ 1][0]), Bbase + (kt + 1) * 32,
                         (unsigned)K);
        __builtin_amdgcn_s_wait_tensorcnt(2);  // in-order: tile kt complete
      } else {
        __builtin_amdgcn_s_wait_tensorcnt(0);
      }
    }
    __syncthreads();  // front buffer visible to all waves

    v16bf af[4], bfv[2];
#pragma unroll
    for (int mt = 0; mt < 4; ++mt) {  // A frag: lanes<16 K{0..7,16..23}, else {8..15,24..31}
      int row = wm * 64 + mt * 16 + lr;
      bf8v lo = *(const bf8v*)&lA[buf][row * 40 + hx * 8];
      bf8v hi = *(const bf8v*)&lA[buf][row * 40 + hx * 8 + 16];
      af[mt] = combine16(lo, hi);
    }
#pragma unroll
    for (int nt = 0; nt < 2; ++nt) {  // B frag: lanes<16 K{0..7,16..23}, else {8..15,24..31}
      int row = wn * 32 + nt * 16 + lr;
      bf8v lo = *(const bf8v*)&lB[buf][row * 40 + hx * 8];
      bf8v hi = *(const bf8v*)&lB[buf][row * 40 + hx * 8 + 16];
      bfv[nt] = combine16(lo, hi);
    }
#pragma unroll
    for (int mt = 0; mt < 4; ++mt)
#pragma unroll
      for (int nt = 0; nt < 2; ++nt)
        acc[mt][nt] = wmma_bf16(af[mt], bfv[nt], acc[mt][nt]);
  }

  // epilogue (C layout: lanes 0-15 -> M=r, lanes 16-31 -> M=8+r; N = lane%16)
  const int mb0 = blockIdx.y * 128 + wm * 64;
  const int nb0 = blockIdx.x * 128 + wn * 32;
#pragma unroll
  for (int nt = 0; nt < 2; ++nt) {
    const int n = nb0 + nt * 16 + lr;
    const float bsv = bias[n];
#pragma unroll
    for (int mt = 0; mt < 4; ++mt) {
      const int mbase = mb0 + mt * 16 + hx * 8;
      if (EPI == 0) {
#pragma unroll
        for (int r = 0; r < 8; ++r)
          Cf[(size_t)(mbase + r) * N + n] = acc[mt][nt][r] + bsv;
      } else if (EPI == 1) {
#pragma unroll
        for (int r = 0; r < 8; ++r)
          Cb[(size_t)(mbase + r) * N + n] = (__bf16)(acc[mt][nt][r] + bsv);
      } else if (EPI == 2) {
        // transposed: out[(b*N + n)*t_ld + s], 8 consecutive M -> one 16B store
        bf8v t;
#pragma unroll
        for (int r = 0; r < 8; ++r) t[r] = (__bf16)(acc[mt][nt][r] + bsv);
        const int bi = mbase / t_ld;
        const int s0 = mbase % t_ld;
        *(bf8v*)&Cb[((size_t)bi * N + n) * t_ld + s0] = t;
      } else {
#pragma unroll
        for (int r = 0; r < 8; ++r) {
          float xv = acc[mt][nt][r] + bsv;
          float gl = 0.5f * xv *
                     (1.0f + tanhf(0.7978845608028654f *
                                   (xv + 0.044715f * xv * xv * xv)));
          Cb[(size_t)(mbase + r) * N + n] = (__bf16)gl;
        }
      }
    }
  }
}

// ---------------------------------------------------------------------------
// Fused sliding-window attention (flash-style online softmax).
// grid = (NBLK, BATCH*NHEAD), 256 threads = 8 waves, each wave owns 32 queries.
// Q,K row-major bf16 [b*S, D]; Vt bf16 [b*D, S] (= V^T per head); O bf16 row-major.
// ---------------------------------------------------------------------------
__global__ __launch_bounds__(256) void k_attn(const __bf16* __restrict__ Q,
                                              const __bf16* __restrict__ Kb,
                                              const __bf16* __restrict__ Vt,
                                              __bf16* __restrict__ O) {
  __shared__ __bf16 pbuf[8][32 * 72];  // per-wave P staging (32 x 64, pitch 72)
  const int blk = blockIdx.x;
  const int bh  = blockIdx.y;
  const int b = bh >> 4, h = bh & 15;
  const int lane = threadIdx.x & 31, wave = threadIdx.x >> 5;
  const int lr = lane & 15, hx = lane >> 4;

  const size_t qrow0 = (size_t)b * SEQ + (size_t)blk * WIN + wave * 32;

  // preload Q A-fragments: 2 m-tiles x 2 k-chunks of 32 over HD=64
  v16bf qf[2][2];
#pragma unroll
  for (int mt = 0; mt < 2; ++mt)
#pragma unroll
    for (int kc = 0; kc < 2; ++kc) {
      const __bf16* qp =
          Q + (qrow0 + mt * 16 + lr) * DMODEL + h * HEADD + kc * 32 + hx * 8;
      bf8v lo = *(const bf8v*)qp;
      bf8v hi = *(const bf8v*)(qp + 16);
      qf[mt][kc] = combine16(lo, hi);
    }

  v8f oacc[2][4];
#pragma unroll
  for (int mt = 0; mt < 2; ++mt)
#pragma unroll
    for (int nt = 0; nt < 4; ++nt) oacc[mt][nt] = v8_splat(0.f);
  v8f rmax[2] = {v8_splat(-1e9f), v8_splat(-1e9f)};
  v8f rsum[2] = {v8_splat(0.f), v8_splat(0.f)};

  // 64-key chunks; halo chunks outside [0,S) are skipped entirely.
  const int cstart = (blk == 0) ? 4 : 0;
  const int cend   = (blk == NBLK - 1) ? 8 : 12;
  const int qp0 = blk * WIN + wave * 32;

  for (int c = cstart; c < cend; ++c) {
    const int k0 = blk * WIN - WIN + c * 64;

    if (c + 1 < cend) {  // gfx1250 global_prefetch of next key chunk
      __builtin_prefetch(
          Kb + ((size_t)b * SEQ + k0 + 64 + lane * 2) * DMODEL + h * HEADD, 0, 3);
    }

    // ---- scores S = Q @ K^T -------------------------------------------------
    v8f sacc[2][4];
#pragma unroll
    for (int mt = 0; mt < 2; ++mt)
#pragma unroll
      for (int nt = 0; nt < 4; ++nt) sacc[mt][nt] = v8_splat(0.f);

#pragma unroll
    for (int kc = 0; kc < 2; ++kc) {
#pragma unroll
      for (int nt = 0; nt < 4; ++nt) {
        const __bf16* kp = Kb + ((size_t)b * SEQ + k0 + nt * 16 + lr) * DMODEL +
                           h * HEADD + kc * 32 + hx * 8;
        bf8v lo = *(const bf8v*)kp;
        bf8v hi = *(const bf8v*)(kp + 16);
        v16bf kf = combine16(lo, hi);
#pragma unroll
        for (int mt = 0; mt < 2; ++mt)
          sacc[mt][nt] = wmma_bf16(qf[mt][kc], kf, sacc[mt][nt]);
      }
    }

    // ---- scale + band mask --------------------------------------------------
#pragma unroll
    for (int mt = 0; mt < 2; ++mt)
#pragma unroll
      for (int nt = 0; nt < 4; ++nt) {
        const int kpos = k0 + nt * 16 + lr;
#pragma unroll
        for (int r = 0; r < 8; ++r) {
          const int qpos = qp0 + mt * 16 + hx * 8 + r;
          const int d = qpos - kpos;
          const float s = sacc[mt][nt][r] * 0.125f;  // 1/sqrt(64)
          sacc[mt][nt][r] = (d <= WIN && d >= -WIN) ? s : -1e9f;
        }
      }

    // ---- online softmax update ---------------------------------------------
#pragma unroll
    for (int mt = 0; mt < 2; ++mt) {
      v8f cmax = sacc[mt][0];
#pragma unroll
      for (int nt = 1; nt < 4; ++nt) cmax = v8_max(cmax, sacc[mt][nt]);
      cmax = v8_half_reduce_max(cmax);
      v8f nm = v8_max(rmax[mt], cmax);
      v8f fac;
#pragma unroll
      for (int r = 0; r < 8; ++r) fac[r] = __expf(rmax[mt][r] - nm[r]);
      rmax[mt] = nm;
#pragma unroll
      for (int nt = 0; nt < 4; ++nt) oacc[mt][nt] *= fac;

      v8f csum = v8_splat(0.f);
#pragma unroll
      for (int nt = 0; nt < 4; ++nt) {
        v8f p;
#pragma unroll
        for (int r = 0; r < 8; ++r) p[r] = __expf(sacc[mt][nt][r] - nm[r]);
        csum += p;
        const int rowb = mt * 16 + hx * 8;
        const int col  = nt * 16 + lr;
#pragma unroll
        for (int r = 0; r < 8; ++r)
          pbuf[wave][(rowb + r) * 72 + col] = (__bf16)p[r];
      }
      csum = v8_half_reduce_add(csum);
      rsum[mt] = rsum[mt] * fac + csum;
    }
    __syncthreads();

    // ---- O += P @ V  (B operand from V^T: K=keys contiguous) ---------------
#pragma unroll
    for (int kc = 0; kc < 2; ++kc) {
      v16bf pf[2];
#pragma unroll
      for (int mt = 0; mt < 2; ++mt) {
        bf8v lo = *(const bf8v*)&pbuf[wave][(mt * 16 + lr) * 72 + kc * 32 + hx * 8];
        bf8v hi =
            *(const bf8v*)&pbuf[wave][(mt * 16 + lr) * 72 + kc * 32 + hx * 8 + 16];
        pf[mt] = combine16(lo, hi);
      }
#pragma unroll
      for (int nt = 0; nt < 4; ++nt) {
        const __bf16* vp = Vt +
                           ((size_t)b * DMODEL + h * HEADD + nt * 16 + lr) * SEQ +
                           k0 + kc * 32 + hx * 8;
        bf8v lo = *(const bf8v*)vp;
        bf8v hi = *(const bf8v*)(vp + 16);
        v16bf vf = combine16(lo, hi);
#pragma unroll
        for (int mt = 0; mt < 2; ++mt)
          oacc[mt][nt] = wmma_bf16(pf[mt], vf, oacc[mt][nt]);
      }
    }
    __syncthreads();
  }

  // ---- normalize and store -------------------------------------------------
#pragma unroll
  for (int mt = 0; mt < 2; ++mt) {
    v8f inv;
#pragma unroll
    for (int r = 0; r < 8; ++r) inv[r] = 1.0f / rsum[mt][r];
#pragma unroll
    for (int nt = 0; nt < 4; ++nt)
#pragma unroll
      for (int r = 0; r < 8; ++r)
        O[(qrow0 + mt * 16 + hx * 8 + r) * DMODEL + h * HEADD + nt * 16 + lr] =
            (__bf16)(oacc[mt][nt][r] * inv[r]);
  }
}

// ---------------------------------------------------------------------------
// out = LayerNorm(a + res) * g + b   (D = 1024, one row per block)
// Writes f32 (outf) and optionally bf16 (outb) for the next GEMM A-operand.
// ---------------------------------------------------------------------------
__global__ __launch_bounds__(256) void k_add_ln(const float* __restrict__ a,
                                                const float* __restrict__ res,
                                                const float* __restrict__ g,
                                                const float* __restrict__ bb,
                                                float* __restrict__ outf,
                                                __bf16* __restrict__ outb) {
  __shared__ float ssum[8], ssq[8];
  const int row = blockIdx.x, tid = threadIdx.x;
  const size_t base = (size_t)row * DMODEL + tid * 4;
  float4 av = *(const float4*)(a + base);
  float4 rv = *(const float4*)(res + base);
  float x0 = av.x + rv.x, x1 = av.y + rv.y, x2 = av.z + rv.z, x3 = av.w + rv.w;
  float s = x0 + x1 + x2 + x3;
  float q = x0 * x0 + x1 * x1 + x2 * x2 + x3 * x3;
#pragma unroll
  for (int m = 1; m < 32; m <<= 1) {
    s += __shfl_xor(s, m, 32);
    q += __shfl_xor(q, m, 32);
  }
  if ((tid & 31) == 0) { ssum[tid >> 5] = s; ssq[tid >> 5] = q; }
  __syncthreads();
  float ts = 0.f, tq = 0.f;
#pragma unroll
  for (int i = 0; i < 8; ++i) { ts += ssum[i]; tq += ssq[i]; }
  const float mu  = ts * (1.0f / DMODEL);
  const float var = tq * (1.0f / DMODEL) - mu * mu;
  const float ri  = rsqrtf(var + 1e-5f);
  float4 gv = *(const float4*)(g + tid * 4);
  float4 bv = *(const float4*)(bb + tid * 4);
  float y0 = (x0 - mu) * ri * gv.x + bv.x;
  float y1 = (x1 - mu) * ri * gv.y + bv.y;
  float y2 = (x2 - mu) * ri * gv.z + bv.z;
  float y3 = (x3 - mu) * ri * gv.w + bv.w;
  if (outf) {
    float4 o; o.x = y0; o.y = y1; o.z = y2; o.w = y3;
    *(float4*)(outf + base) = o;
  }
  if (outb) {
    bf4v o;
    o[0] = (__bf16)y0; o[1] = (__bf16)y1; o[2] = (__bf16)y2; o[3] = (__bf16)y3;
    *(bf4v*)(outb + base) = o;
  }
}

// ---------------------------------------------------------------------------
// launch
// ---------------------------------------------------------------------------
extern "C" void kernel_launch(void* const* d_in, const int* in_sizes, int n_in,
                              void* d_out, int out_size, void* d_ws,
                              size_t ws_size, hipStream_t stream) {
  (void)in_sizes; (void)n_in; (void)out_size; (void)ws_size;
  const float* x  = (const float*)d_in[0];
  const float* Wq = (const float*)d_in[1];
  const float* bq = (const float*)d_in[2];
  const float* Wk = (const float*)d_in[3];
  const float* bk = (const float*)d_in[4];
  const float* Wv = (const float*)d_in[5];
  const float* bv = (const float*)d_in[6];
  const float* Wo = (const float*)d_in[7];
  const float* bo = (const float*)d_in[8];
  const float* g1 = (const float*)d_in[9];
  const float* b1 = (const float*)d_in[10];
  const float* W1 = (const float*)d_in[11];
  const float* c1 = (const float*)d_in[12];
  const float* W2 = (const float*)d_in[13];
  const float* c2 = (const float*)d_in[14];
  const float* g2 = (const float*)d_in[15];
  const float* b2 = (const float*)d_in[16];

  char* ws = (char*)d_ws;
  size_t off = 0;
  auto take = [&](size_t bytes) -> char* {
    char* p = ws + off;
    off += (bytes + 255) & ~(size_t)255;
    return p;
  };
  __bf16* XB  = (__bf16*)take((size_t)MROWS * DMODEL * 2);
  __bf16* WQT = (__bf16*)take((size_t)DMODEL * DMODEL * 2);
  __bf16* WKT = (__bf16*)take((size_t)DMODEL * DMODEL * 2);
  __bf16* WVT = (__bf16*)take((size_t)DMODEL * DMODEL * 2);
  __bf16* WOT = (__bf16*)take((size_t)DMODEL * DMODEL * 2);
  __bf16* W1T = (__bf16*)take((size_t)FFNDIM * DMODEL * 2);
  __bf16* W2T = (__bf16*)take((size_t)DMODEL * FFNDIM * 2);
  __bf16* QB  = (__bf16*)take((size_t)MROWS * DMODEL * 2);
  __bf16* KB  = (__bf16*)take((size_t)MROWS * DMODEL * 2);
  __bf16* VTb = (__bf16*)take((size_t)BATCH * DMODEL * SEQ * 2);
  __bf16* AT  = (__bf16*)take((size_t)MROWS * DMODEL * 2);
  float*  TMPF= (float*)take((size_t)MROWS * DMODEL * 4);
  float*  HF  = (float*)take((size_t)MROWS * DMODEL * 4);
  __bf16* HB  = (__bf16*)take((size_t)MROWS * DMODEL * 2);
  __bf16* F1B = (__bf16*)take((size_t)MROWS * FFNDIM * 2);

  const dim3 t256(256);
  const dim3 t32x8(32, 8);

  // 1) convert activations / weights to bf16 (weights transposed to NxK)
  k_cvt_bf16<<<(MROWS * DMODEL / 4 + 255) / 256, t256, 0, stream>>>(
      x, XB, MROWS * DMODEL / 4);
  k_transpose_cvt<<<dim3(DMODEL / 32, DMODEL / 32), t32x8, 0, stream>>>(Wq, WQT, DMODEL, DMODEL);
  k_transpose_cvt<<<dim3(DMODEL / 32, DMODEL / 32), t32x8, 0, stream>>>(Wk, WKT, DMODEL, DMODEL);
  k_transpose_cvt<<<dim3(DMODEL / 32, DMODEL / 32), t32x8, 0, stream>>>(Wv, WVT, DMODEL, DMODEL);
  k_transpose_cvt<<<dim3(DMODEL / 32, DMODEL / 32), t32x8, 0, stream>>>(Wo, WOT, DMODEL, DMODEL);
  k_transpose_cvt<<<dim3(FFNDIM / 32, DMODEL / 32), t32x8, 0, stream>>>(W1, W1T, DMODEL, FFNDIM);
  k_transpose_cvt<<<dim3(DMODEL / 32, FFNDIM / 32), t32x8, 0, stream>>>(W2, W2T, FFNDIM, DMODEL);

  // 2) projections (V written transposed per head for the PV WMMA B-operand)
  const dim3 gD(DMODEL / 128, MROWS / 128);
  k_gemm<1><<<gD, t256, 0, stream>>>(XB, WQT, bq, nullptr, QB, MROWS, DMODEL, DMODEL, 1);
  k_gemm<1><<<gD, t256, 0, stream>>>(XB, WKT, bk, nullptr, KB, MROWS, DMODEL, DMODEL, 1);
  k_gemm<2><<<gD, t256, 0, stream>>>(XB, WVT, bv, nullptr, VTb, MROWS, DMODEL, DMODEL, SEQ);

  // 3) fused sliding-window attention
  k_attn<<<dim3(NBLK, BATCH * NHEAD), t256, 0, stream>>>(QB, KB, VTb, AT);

  // 4) output projection + residual LN -> h (f32 + bf16)
  k_gemm<0><<<gD, t256, 0, stream>>>(AT, WOT, bo, TMPF, nullptr, MROWS, DMODEL, DMODEL, 1);
  k_add_ln<<<MROWS, t256, 0, stream>>>(TMPF, x, g1, b1, HF, HB);

  // 5) FFN (gelu fused into first GEMM epilogue) + residual LN -> output
  k_gemm<3><<<dim3(FFNDIM / 128, MROWS / 128), t256, 0, stream>>>(
      HB, W1T, c1, nullptr, F1B, MROWS, FFNDIM, DMODEL, 1);
  k_gemm<0><<<gD, t256, 0, stream>>>(F1B, W2T, c2, TMPF, nullptr, MROWS, DMODEL, FFNDIM, 1);
  k_add_ln<<<MROWS, t256, 0, stream>>>(TMPF, HF, g2, b2, (float*)d_out, nullptr);
}
